// DeepCNF_11751030522669
// MI455X (gfx1250) — matrix operand order repaired and checked by reading, not compiled, not run on hardware
//
#include <hip/hip_runtime.h>
#include <hip/hip_bf16.h>
#include <math.h>

// ---------------- problem constants ----------------
constexpr int Bn    = 128;
constexpr int Tn    = 1024;
constexpr int INC   = 42;
constexpr int HID   = 256;
constexpr int KW    = 11;
constexpr int NTAG  = 8;
constexpr int PADW  = KW / 2;           // 5
constexpr int TP    = Tn + 2 * PADW;    // 1034  (time-padded)
constexpr int C0    = 64;               // layer-0 input channels padded 42 -> 64
constexpr int TBLK  = 64;               // time steps per block
constexpr int TROWS = TBLK + 2 * PADW;  // 74 LDS halo rows

// ---------------- types ----------------
typedef __bf16 bf16_t;
typedef __attribute__((ext_vector_type(16))) bf16_t v16bf;
typedef __attribute__((ext_vector_type(8)))  float  v8f;

union FragBF {            // one WMMA bf16 operand (16 bf16 per lane = 32B)
    v16bf v;
    uint4 q[2];
};
union Pack8 {             // 8 bf16 packed for a 16B store
    unsigned short s[8];
    uint4 q;
};

__device__ __forceinline__ unsigned short f2bf(float f) {
    unsigned u = __float_as_uint(f);
    u += 0x7FFFu + ((u >> 16) & 1u);    // round-to-nearest-even
    return (unsigned short)(u >> 16);
}
__device__ __forceinline__ float bf2f(unsigned short h) {
    return __uint_as_float(((unsigned)h) << 16);
}

// ---------------- workspace zeroing ----------------
__global__ void zero_kernel(uint4* p, long long n16) {
    long long i = (long long)blockIdx.x * blockDim.x + threadIdx.x;
    uint4 z; z.x = z.y = z.z = z.w = 0u;
    if (i < n16) p[i] = z;
}

// ---------------- pack x: (B,T,42) f32 -> (B,TP,64) bf16, pad zeros ----------------
__global__ void pack_x_kernel(const float* __restrict__ x, unsigned short* __restrict__ x0) {
    int idx = blockIdx.x * blockDim.x + threadIdx.x;   // b*T + t
    if (idx >= Bn * Tn) return;
    int b = idx / Tn, t = idx % Tn;
    const float* src = x + (size_t)idx * INC;
    unsigned short* dst = x0 + ((size_t)b * TP + (t + PADW)) * C0;
    #pragma unroll
    for (int c = 0; c < C0; ++c)
        dst[c] = (c < INC) ? f2bf(src[c]) : (unsigned short)0;
}

// ---------------- pack weights: (CO,CIs,KW) f32 -> (KW,CO,CIp) bf16 ----------------
__global__ void pack_w_kernel(const float* __restrict__ w, unsigned short* __restrict__ wp,
                              int CO, int CIs, int CIp) {
    int idx = blockIdx.x * blockDim.x + threadIdx.x;
    int total = KW * CO * CIp;
    if (idx >= total) return;
    int cip = idx % CIp;
    int co  = (idx / CIp) % CO;
    int k   = idx / (CIp * CO);
    float v = (cip < CIs) ? w[(size_t)co * CIs * KW + (size_t)cip * KW + k] : 0.0f;
    wp[idx] = f2bf(v);
}

// ---------------- conv layer via WMMA bf16, LDS-staged activations ----------------
// in : (B, TP, CIN) bf16   out : (B, TP, HID) bf16 (valid region [PADW, PADW+T))
// wp : (KW, HID, CIN) bf16
// block: 8 waves = 256 out-channels x 64 time-steps of one batch sample
// wave : 32 out-channels x 64 time-steps (8 f32 accumulators)
template <int CIN>
__global__ __launch_bounds__(256)
void conv_wmma_kernel(const unsigned short* __restrict__ in,
                      unsigned short* __restrict__ out,
                      const unsigned short* __restrict__ wp,
                      const float* __restrict__ bias) {
    __shared__ __align__(16) unsigned short lds[TROWS * CIN];

    const int tid    = threadIdx.x;
    const int lane   = tid & 31;
    const int wid    = tid >> 5;
    const int b      = blockIdx.x >> 4;            // T/TBLK = 16 time blocks
    const int t0     = (blockIdx.x & 15) * TBLK;
    const int co0    = wid * 32;
    const int laneLo = lane & 15;
    const int laneHi = lane >> 4;

    // ---- cooperative stage of the 74-row halo tile into LDS ----
    {
        const uint4* gsrc = (const uint4*)(in + ((size_t)b * TP + t0) * CIN);
        uint4* ldst = (uint4*)lds;
        const int n16 = (TROWS * CIN) / 8;         // uint4 count
        for (int i = tid; i < n16; i += 256)
            ldst[i] = gsrc[i];
    }
    __syncthreads();

    v8f acc[2][4] = {};   // [co-subtile][t-subtile]

    for (int k = 0; k < KW; ++k) {
        const unsigned short* wk = wp + (size_t)k * HID * CIN;
        // warm next tap's weight rows for this wave (global_prefetch_b8)
        if (k + 1 < KW)
            __builtin_prefetch(wk + (size_t)HID * CIN + (size_t)(co0 + laneLo) * CIN, 0, 1);
        #pragma unroll
        for (int ci0 = 0; ci0 < CIN; ci0 += 32) {
            // A fragments (weights) from global: 16x32, K halves {kb..kb+7, kb+16..kb+23}
            FragBF a0, a1;
            {
                const unsigned short* p = wk + (size_t)(co0 + laneLo) * CIN + ci0 + laneHi * 8;
                a0.q[0] = *(const uint4*)(p);
                a0.q[1] = *(const uint4*)(p + 16);
                const unsigned short* p1 = p + 16 * CIN;
                a1.q[0] = *(const uint4*)(p1);
                a1.q[1] = *(const uint4*)(p1 + 16);
            }
            // B fragments (activations) from LDS, one per t-subtile, reused by both co tiles
            #pragma unroll
            for (int ts = 0; ts < 4; ++ts) {
                FragBF bf;
                const int row = ts * 16 + laneLo + k;           // local padded time row
                const unsigned short* p = lds + row * CIN + ci0 + laneHi * 8;
                bf.q[0] = *(const uint4*)(p);
                bf.q[1] = *(const uint4*)(p + 16);
                acc[0][ts] = __builtin_amdgcn_wmma_f32_16x16x32_bf16(false, a0.v, false, bf.v, (short)0, acc[0][ts], false, false);
                acc[1][ts] = __builtin_amdgcn_wmma_f32_16x16x32_bf16(false, a1.v, false, bf.v, (short)0, acc[1][ts], false, false);
            }
        }
    }

    // epilogue: bias + tanh + bf16 pack; each lane stores 8 consecutive channels
    #pragma unroll
    for (int cs = 0; cs < 2; ++cs) {
        #pragma unroll
        for (int ts = 0; ts < 4; ++ts) {
            const int coB = co0 + cs * 16 + laneHi * 8;
            const int tt  = t0 + ts * 16 + laneLo;
            Pack8 pk;
            #pragma unroll
            for (int v = 0; v < 8; ++v) {
                float f = acc[cs][ts][v] + bias[coB + v];
                pk.s[v] = f2bf(tanhf(f));
            }
            *(uint4*)(out + ((size_t)b * TP + (PADW + tt)) * HID + coB) = pk.q;
        }
    }
}

// ---------------- 1x1 output conv: emissions (B,T,NTAG) f32 ----------------
__global__ void emission_kernel(const unsigned short* __restrict__ h,
                                const float* __restrict__ w_out,
                                const float* __restrict__ b_out,
                                float* __restrict__ em) {
    int idx = blockIdx.x * blockDim.x + threadIdx.x;   // b*T + t
    if (idx >= Bn * Tn) return;
    int b = idx / Tn, t = idx % Tn;
    const unsigned short* hp = h + ((size_t)b * TP + (PADW + t)) * HID;
    float acc[NTAG];
    #pragma unroll
    for (int j = 0; j < NTAG; ++j) acc[j] = b_out[j];
    for (int ci = 0; ci < HID; ++ci) {
        float hv = bf2f(hp[ci]);
        #pragma unroll
        for (int j = 0; j < NTAG; ++j)
            acc[j] += hv * w_out[j * HID + ci];
    }
    float* o = em + (size_t)idx * NTAG;
    #pragma unroll
    for (int j = 0; j < NTAG; ++j) o[j] = acc[j];
}

// ---------------- CRF NLL: one wave per sequence ----------------
__global__ __launch_bounds__(32)
void crf_kernel(const float* __restrict__ em, const int* __restrict__ tags,
                const int* __restrict__ lengths,
                const float* __restrict__ start_trans,
                const float* __restrict__ end_trans,
                const float* __restrict__ trans,
                float* __restrict__ nll) {
    const int b    = blockIdx.x;
    const int lane = threadIdx.x;
    const int len  = lengths[b];
    const float* emb = em + (size_t)b * Tn * NTAG;
    const int*   tg  = tags + (size_t)b * Tn;

    // lane j (j<8) holds column j of the transition matrix
    float trCol[NTAG];
    #pragma unroll
    for (int i = 0; i < NTAG; ++i)
        trCol[i] = (lane < NTAG) ? trans[i * NTAG + lane] : 0.0f;

    float alpha = (lane < NTAG) ? (start_trans[lane] + emb[lane]) : -1e30f;

    // gold-path score (lane 0 mirrors the scan)
    int   prev  = tg[0];
    float score = start_trans[prev] + emb[prev];

    for (int t = 1; t < Tn; ++t) {
        const bool valid = (t < len);
        float emt = (lane < NTAG) ? emb[t * NTAG + lane] : 0.0f;
        // logsumexp_i( alpha_i + trans[i][lane] )
        float vals[NTAG];
        float m = -1e30f;
        #pragma unroll
        for (int i = 0; i < NTAG; ++i) {
            float av = __shfl(alpha, i);
            vals[i] = av + trCol[i];
            m = fmaxf(m, vals[i]);
        }
        float s = 0.0f;
        #pragma unroll
        for (int i = 0; i < NTAG; ++i) s += expf(vals[i] - m);
        float nxt = m + logf(s) + emt;
        if (valid && lane < NTAG) alpha = nxt;
        if (valid && lane == 0) {
            int ct = tg[t];
            score += trans[prev * NTAG + ct] + emb[t * NTAG + ct];
            prev = ct;
        }
    }

    // logZ = lse_j( alpha_j + end_trans[j] )
    float term = (lane < NTAG) ? (alpha + end_trans[lane]) : -1e30f;
    float vals[NTAG];
    float m = -1e30f;
    #pragma unroll
    for (int i = 0; i < NTAG; ++i) {
        vals[i] = __shfl(term, i);
        m = fmaxf(m, vals[i]);
    }
    float s = 0.0f;
    #pragma unroll
    for (int i = 0; i < NTAG; ++i) s += expf(vals[i] - m);
    float logz = m + logf(s);

    // broadcast last gold tag from lane 0
    int lastTag = __shfl(prev, 0);
    float sc    = __shfl(score, 0);
    if (lane == 0)
        nll[b] = logz - (sc + end_trans[lastTag]);
}

__global__ void reduce_kernel(const float* __restrict__ nll, float* __restrict__ out) {
    if (threadIdx.x == 0 && blockIdx.x == 0) {
        float s = 0.0f;
        for (int i = 0; i < Bn; ++i) s += nll[i];
        out[0] = s;
    }
}

// ---------------- launcher ----------------
extern "C" void kernel_launch(void* const* d_in, const int* in_sizes, int n_in,
                              void* d_out, int out_size, void* d_ws, size_t ws_size,
                              hipStream_t stream) {
    (void)in_sizes; (void)n_in; (void)out_size; (void)ws_size;

    const float* x        = (const float*)d_in[0];
    const int*   lengths  = (const int*)d_in[1];
    const int*   tags     = (const int*)d_in[3];
    const float* wL[5]    = {(const float*)d_in[4], (const float*)d_in[6], (const float*)d_in[8],
                             (const float*)d_in[10], (const float*)d_in[12]};
    const float* bL[5]    = {(const float*)d_in[5], (const float*)d_in[7], (const float*)d_in[9],
                             (const float*)d_in[11], (const float*)d_in[13]};
    const float* w_out    = (const float*)d_in[14];
    const float* b_out    = (const float*)d_in[15];
    const float* st_tr    = (const float*)d_in[16];
    const float* en_tr    = (const float*)d_in[17];
    const float* trans    = (const float*)d_in[18];

    // ---- workspace layout (elements) ----
    const size_t X0E = (size_t)Bn * TP * C0;       // bf16
    const size_t HE  = (size_t)Bn * TP * HID;      // bf16
    const size_t WP0 = (size_t)KW * HID * C0;      // bf16
    const size_t WPH = (size_t)KW * HID * HID;     // bf16
    const size_t WPE = WP0 + 4 * WPH;

    unsigned short* x0 = (unsigned short*)d_ws;
    unsigned short* hA = x0 + X0E;
    unsigned short* hB = hA + HE;
    unsigned short* wp = hB + HE;
    unsigned short* wpL[5] = {wp, wp + WP0, wp + WP0 + WPH, wp + WP0 + 2 * WPH, wp + WP0 + 3 * WPH};
    float* em  = (float*)(wp + WPE);
    float* nll = em + (size_t)Bn * Tn * NTAG;

    // 1. zero activation buffers (covers time-pad halo rows)
    {
        long long zbytes = (long long)(X0E + 2 * HE) * 2;
        long long n16 = zbytes / 16;
        int blocks = (int)((n16 + 255) / 256);
        zero_kernel<<<blocks, 256, 0, stream>>>((uint4*)d_ws, n16);
    }
    // 2. pack input
    pack_x_kernel<<<(Bn * Tn + 255) / 256, 256, 0, stream>>>(x, x0);
    // 3. pack weights
    pack_w_kernel<<<(int)((WP0 + 255) / 256), 256, 0, stream>>>(wL[0], wpL[0], HID, INC, C0);
    for (int l = 1; l < 5; ++l)
        pack_w_kernel<<<(int)((WPH + 255) / 256), 256, 0, stream>>>(wL[l], wpL[l], HID, HID, HID);

    // 4. conv stack (ping-pong); one block per (batch, 64-step strip)
    const int convBlocks = Bn * (Tn / TBLK);       // 2048
    conv_wmma_kernel<C0> <<<convBlocks, 256, 0, stream>>>(x0, hA, wpL[0], bL[0]);
    conv_wmma_kernel<HID><<<convBlocks, 256, 0, stream>>>(hA, hB, wpL[1], bL[1]);
    conv_wmma_kernel<HID><<<convBlocks, 256, 0, stream>>>(hB, hA, wpL[2], bL[2]);
    conv_wmma_kernel<HID><<<convBlocks, 256, 0, stream>>>(hA, hB, wpL[3], bL[3]);
    conv_wmma_kernel<HID><<<convBlocks, 256, 0, stream>>>(hB, hA, wpL[4], bL[4]);

    // 5. emissions
    emission_kernel<<<(Bn * Tn + 255) / 256, 256, 0, stream>>>(hA, w_out, b_out, em);

    // 6. CRF NLL per sequence, then deterministic reduction
    crf_kernel<<<Bn, 32, 0, stream>>>(em, tags, lengths, st_tr, en_tr, trans, nll);
    reduce_kernel<<<1, 32, 0, stream>>>(nll, (float*)d_out);
}
